// CCPGraph_63977832841258
// MI455X (gfx1250) — matrix operand order repaired and checked by reading, not compiled, not run on hardware
//
#include <hip/hip_runtime.h>
#include <hip/hip_bf16.h>

typedef __attribute__((ext_vector_type(16))) _Float16 v16h;
typedef __attribute__((ext_vector_type(8)))  _Float16 v8h;
typedef __attribute__((ext_vector_type(8)))  float    v8f;

__device__ __forceinline__ v16h pack16(v8h lo, v8h hi) {
  v16h a;
#pragma unroll
  for (int i = 0; i < 8; ++i) { a[i] = lo[i]; a[i + 8] = hi[i]; }
  return a;
}

__device__ __forceinline__ v16h loadB(const _Float16* bp) {
  return pack16(*(const v8h*)bp, *(const v8h*)(bp + 16));
}

__device__ __forceinline__ v8h cvt8(float4 a, float4 b) {
  v8h r;
  r[0] = (_Float16)a.x; r[1] = (_Float16)a.y; r[2] = (_Float16)a.z; r[3] = (_Float16)a.w;
  r[4] = (_Float16)b.x; r[5] = (_Float16)b.y; r[6] = (_Float16)b.z; r[7] = (_Float16)b.w;
  return r;
}

__device__ __forceinline__ v8f wmma16x16x32(v16h a, v16h b, v8f c) {
  // (neg_a, A, neg_b, B, c_mod, C, reuse_a, reuse_b)
  return __builtin_amdgcn_wmma_f32_16x16x32_f16(false, a, false, b, (short)0, c,
                                                false, false);
}

__device__ __forceinline__ v8f v8f_zero() {
  v8f z;
#pragma unroll
  for (int i = 0; i < 8; ++i) z[i] = 0.0f;
  return z;
}

// Load a 16-row A fragment chunk (K-offset koff) straight from a [.,64] f16/f32
// matrix: lane covers row (lane&15), halves {koff..koff+7} and {koff+16..koff+23}.
template <bool FEAT_F16>
__device__ __forceinline__ v16h loadA64(const void* feat, long row, int koff) {
  if constexpr (FEAT_F16) {
    const _Float16* fp = (const _Float16*)feat + row * 64 + koff;
    return pack16(*(const v8h*)fp, *(const v8h*)(fp + 16));
  } else {
    const float* fp = (const float*)feat + row * 64 + koff;
    float4 q0 = *(const float4*)(fp);
    float4 q1 = *(const float4*)(fp + 4);
    float4 q2 = *(const float4*)(fp + 16);
    float4 q3 = *(const float4*)(fp + 20);
    return pack16(cvt8(q0, q1), cvt8(q2, q3));
  }
}

__device__ __forceinline__ void atomicMaxF(float* addr, float v) {
  if (v >= 0.0f) atomicMax((int*)addr, __float_as_int(v));
  else           atomicMin((unsigned int*)addr, (unsigned int)__float_as_int(v));
}

// ---------------------------------------------------------------- embedding
__global__ void k_embed(const int* __restrict__ x, const float* __restrict__ embeds,
                        _Float16* __restrict__ h0, int N) {
  int i = blockIdx.x * blockDim.x + threadIdx.x;  // over N*64
  if (i < N * 64) {
    int node = i >> 6, c = i & 63;
    h0[i] = (_Float16)embeds[x[node] * 64 + c];
  }
}

// ------------------------------------------------- root GEMM: tanh(X@W + b)
template <int NOUT, bool FEAT_F16>
__global__ void k_root(const void* __restrict__ feat,
                       const float* __restrict__ W_root,  // [64,NOUT]
                       const float* __restrict__ b_root,
                       float* __restrict__ out, int N) {
  constexpr int K = 64;
  constexpr int NT = NOUT / 16;
  __shared__ __attribute__((aligned(16))) _Float16 sB[NOUT * K];  // [n][k]
  __shared__ float sBias[NOUT];
  int tid = threadIdx.x;
  for (int i = tid; i < NOUT * K; i += blockDim.x) {
    int n = i / K, k = i % K;
    sB[i] = (_Float16)W_root[k * NOUT + n];
  }
  if (tid < NOUT) sBias[tid] = b_root[tid];
  __syncthreads();

  int wave = tid >> 5, lane = tid & 31;
  int half = lane >> 4, col = lane & 15;
  int arow = lane & 15;

  v16h bfrag[2];  // whole B panel in registers when NOUT==16
  if constexpr (NT == 1) {
#pragma unroll
    for (int kk = 0; kk < 2; ++kk)
      bfrag[kk] = loadB(sB + col * K + kk * 32 + half * 8);
  }

  int ntiles = (N + 15) / 16;
  int stride = gridDim.x * 8;
  for (int tile = blockIdx.x * 8 + wave; tile < ntiles; tile += stride) {
    int base = tile * 16;
    long node = min(base + arow, N - 1);
    v8f acc[NT];
#pragma unroll
    for (int t = 0; t < NT; ++t) acc[t] = v8f_zero();

#pragma unroll
    for (int kk = 0; kk < 2; ++kk) {
      v16h a = loadA64<FEAT_F16>(feat, node, kk * 32 + half * 8);
#pragma unroll
      for (int t = 0; t < NT; ++t) {
        v16h b;
        if constexpr (NT == 1) b = bfrag[kk];
        else b = loadB(sB + (t * 16 + col) * K + kk * 32 + half * 8);
        acc[t] = wmma16x16x32(a, b, acc[t]);
      }
    }
#pragma unroll
    for (int t = 0; t < NT; ++t) {
      float bias = sBias[t * 16 + col];
#pragma unroll
      for (int r = 0; r < 8; ++r) {
        int nd = base + r + half * 8;
        if (nd < N) out[(long)nd * NOUT + t * 16 + col] = tanhf(acc[t][r] + bias);
      }
    }
  }
}

// --------------------------- edge GEMM: scatter-add tanh([X[src],attr]@W+b)
// A fragments built directly from the random gather (no LDS staging):
// invalid / padded rows only pollute their own C rows, which are masked at the
// scatter, so no zeroing of garbage A rows is needed.
template <int NOUT, bool FEAT_F16>
__global__ void k_edge(const void* __restrict__ feat,        // [N,64]
                       const int* __restrict__ edge_dst,     // [E]
                       const int* __restrict__ edge_src,     // [E]
                       const float* __restrict__ edge_attr,  // [E,6]
                       const float* __restrict__ W_neg,      // [70,NOUT]
                       const float* __restrict__ b_neg,
                       float* __restrict__ out,              // [N,NOUT] accumulate
                       int E) {
  constexpr int KP = 96;  // 70 padded to 3*32
  constexpr int NT = NOUT / 16;
  __shared__ __attribute__((aligned(16))) _Float16 sB[NOUT * KP];  // [n][k]
  __shared__ float sBias[NOUT];
  int tid = threadIdx.x;
  for (int i = tid; i < NOUT * KP; i += blockDim.x) {
    int n = i / KP, k = i % KP;
    sB[i] = (k < 70) ? (_Float16)W_neg[k * NOUT + n] : (_Float16)0.0f;
  }
  if (tid < NOUT) sBias[tid] = b_neg[tid];
  __syncthreads();

  int wave = tid >> 5, lane = tid & 31;
  int half = lane >> 4, col = lane & 15;
  int arow = lane & 15;

  v16h bfrag[3];  // whole B panel in registers when NOUT==16
  if constexpr (NT == 1) {
#pragma unroll
    for (int kk = 0; kk < 3; ++kk)
      bfrag[kk] = loadB(sB + col * KP + kk * 32 + half * 8);
  }

  int ntiles = (E + 15) / 16;
  int stride = gridDim.x * 8;
  for (int tile = blockIdx.x * 8 + wave; tile < ntiles; tile += stride) {
    int ebase = tile * 16;
    int e = ebase + arow;
    bool valid = e < E;
    long src = valid ? (long)edge_src[e] : 0;

    // prefetch next tile's index words (linear stream)
    int ne = ebase + stride * 16 + arow;
    if (ne < E) {
      __builtin_prefetch(edge_src + ne, 0, 1);
      __builtin_prefetch(edge_dst + ne, 0, 1);
    }

    // A fragments: kk=0,1 from gathered features; kk=2 = [attr(6), zeros]
    v16h afrag[3];
#pragma unroll
    for (int kk = 0; kk < 2; ++kk)
      afrag[kk] = loadA64<FEAT_F16>(feat, src, kk * 32 + half * 8);
    {
      v16h a2;
#pragma unroll
      for (int i = 0; i < 16; ++i) a2[i] = (_Float16)0.0f;
      if (half == 0 && valid) {
        const float* ap = edge_attr + (long)e * 6;
#pragma unroll
        for (int j = 0; j < 6; ++j) a2[j] = (_Float16)ap[j];
      }
      afrag[2] = a2;
    }

    v8f acc[NT];
#pragma unroll
    for (int t = 0; t < NT; ++t) acc[t] = v8f_zero();

#pragma unroll
    for (int kk = 0; kk < 3; ++kk) {
#pragma unroll
      for (int t = 0; t < NT; ++t) {
        v16h b;
        if constexpr (NT == 1) b = bfrag[kk];
        else b = loadB(sB + (t * 16 + col) * KP + kk * 32 + half * 8);
        acc[t] = wmma16x16x32(afrag[kk], b, acc[t]);
      }
    }

#pragma unroll
    for (int t = 0; t < NT; ++t) {
      float bias = sBias[t * 16 + col];
#pragma unroll
      for (int rr = 0; rr < 8; ++rr) {
        int ee = ebase + rr + half * 8;
        if (ee < E) {
          float v = tanhf(acc[t][rr] + bias);
          atomicAdd(out + (long)edge_dst[ee] * NOUT + t * 16 + col, v);
        }
      }
    }
  }
}

// -------------------------------------------------------------- gate MLP
__global__ void k_gate(const float* __restrict__ h2,
                       const float* __restrict__ Wg1, const float* __restrict__ bg1,
                       const float* __restrict__ Wg2, const float* __restrict__ bg2,
                       const float* __restrict__ Wg3, const float* __restrict__ bg3,
                       float* __restrict__ gate, float* __restrict__ gate_total,
                       int N) {
  __shared__ float sW1[16 * 64], sW2[64 * 32], sW3[32], sB1[64], sB2[32];
  int tid = threadIdx.x;
  for (int i = tid; i < 16 * 64; i += blockDim.x) sW1[i] = Wg1[i];
  for (int i = tid; i < 64 * 32; i += blockDim.x) sW2[i] = Wg2[i];
  if (tid < 64) sB1[tid] = bg1[tid];
  if (tid < 32) { sW3[tid] = Wg3[tid]; sB2[tid] = bg2[tid]; }
  __syncthreads();

  int i = blockIdx.x * blockDim.x + tid;
  float g = 0.0f;
  if (i < N) {
    float h[16];
#pragma unroll
    for (int k = 0; k < 16; ++k) h[k] = h2[(long)i * 16 + k];
    float g1[64];
    for (int j = 0; j < 64; ++j) {
      float s = sB1[j];
#pragma unroll
      for (int k = 0; k < 16; ++k) s += h[k] * sW1[k * 64 + j];
      g1[j] = fmaxf(s, 0.0f);
    }
    float o = bg3[0];
    for (int j = 0; j < 32; ++j) {
      float s = sB2[j];
#pragma unroll 8
      for (int k = 0; k < 64; ++k) s += g1[k] * sW2[k * 32 + j];
      o += fmaxf(s, 0.0f) * sW3[j];
    }
    g = o;
    gate[i] = g;
  }
  float s = (i < N) ? g : 0.0f;
#pragma unroll
  for (int off = 16; off > 0; off >>= 1) s += __shfl_xor(s, off, 32);
  if ((tid & 31) == 0) atomicAdd(gate_total, s);
}

// ---------------------------------------------------------- softmax pieces
__global__ void k_init(float* segmax, float* segsum, float* emb, float* gate_total,
                       int G) {
  int i = blockIdx.x * blockDim.x + threadIdx.x;
  if (i < G) { segmax[i] = -__builtin_huge_valf(); segsum[i] = 0.0f; }
  if (i < G * 16) emb[i] = 0.0f;
  if (i == 0) *gate_total = 0.0f;
}

__global__ void k_segmax(const float* __restrict__ gate, const int* __restrict__ batch,
                         float* __restrict__ segmax, int N) {
  int i = blockIdx.x * blockDim.x + threadIdx.x;
  if (i < N) atomicMaxF(&segmax[batch[i]], gate[i]);
}

__global__ void k_exp(const float* __restrict__ gate, const int* __restrict__ batch,
                      const float* __restrict__ segmax, float* __restrict__ expv,
                      float* __restrict__ segsum, const float* __restrict__ gate_total,
                      float* __restrict__ att, int N) {
  int i = blockIdx.x * blockDim.x + threadIdx.x;
  if (i < N) {
    float g = gate[i];
    int b = batch[i];
    float e = expf(g - segmax[b]);
    expv[i] = e;
    atomicAdd(&segsum[b], e);
    att[i] = g - (*gate_total) / (float)N;
  }
}

__global__ void k_scatter(const float* __restrict__ h2, const float* __restrict__ expv,
                          const float* __restrict__ segsum, const int* __restrict__ batch,
                          float* __restrict__ emb, int N) {
  int t = blockIdx.x * blockDim.x + threadIdx.x;  // N*16
  if (t < N * 16) {
    int i = t >> 4, c = t & 15;
    int b = batch[i];
    float w = expv[i] / (segsum[b] + 1e-16f);
    atomicAdd(&emb[b * 16 + c], w * h2[(long)i * 16 + c]);
  }
}

// ------------------------------------------------------------------- head
__global__ void k_final(const float* __restrict__ emb, const float* __restrict__ W_semi,
                        const float* __restrict__ gamma, const float* __restrict__ beta,
                        const float* __restrict__ mean, const float* __restrict__ var,
                        const float* __restrict__ W_final, const float* __restrict__ b_final,
                        float* __restrict__ out, float* __restrict__ sig,
                        float* __restrict__ out1, int G) {
  __shared__ float sW[16 * 200], sScale[200], sShift[200], sWf[200];
  int tid = threadIdx.x;
  for (int i = tid; i < 16 * 200; i += blockDim.x) sW[i] = W_semi[i];
  for (int i = tid; i < 200; i += blockDim.x) {
    float rs = rsqrtf(var[i] + 1e-5f) * gamma[i];
    sScale[i] = rs;
    sShift[i] = beta[i] - mean[i] * rs;
    sWf[i] = W_final[i];
  }
  __syncthreads();
  int g = blockIdx.x * blockDim.x + tid;
  if (g < G) {
    float e[16];
#pragma unroll
    for (int k = 0; k < 16; ++k) e[k] = emb[g * 16 + k];
    float o = b_final[0];
    for (int j = 0; j < 200; ++j) {
      float s = 0.0f;
#pragma unroll
      for (int k = 0; k < 16; ++k) s += e[k] * sW[k * 200 + j];
      float y = s * sScale[j] + sShift[j];
      out1[(long)g * 200 + j] = y;
      o += y * sWf[j];
    }
    out[g] = o;
    sig[g] = 1.0f / (1.0f + expf(-o));
  }
}

extern "C" void kernel_launch(void* const* d_in, const int* in_sizes, int n_in,
                              void* d_out, int out_size, void* d_ws, size_t ws_size,
                              hipStream_t stream) {
  const int*   x          = (const int*)d_in[0];
  const int*   edge_index = (const int*)d_in[1];
  const float* edge_attr  = (const float*)d_in[2];
  const int*   batch      = (const int*)d_in[3];
  // d_in[4] = num_graphs (device scalar, unused: G derived from out_size)
  const float* embeds  = (const float*)d_in[5];
  const float* W_neg1  = (const float*)d_in[6];
  const float* b_neg1  = (const float*)d_in[7];
  const float* W_root1 = (const float*)d_in[8];
  const float* b_root1 = (const float*)d_in[9];
  const float* W_neg2  = (const float*)d_in[10];
  const float* b_neg2  = (const float*)d_in[11];
  const float* W_root2 = (const float*)d_in[12];
  const float* b_root2 = (const float*)d_in[13];
  const float* Wg1 = (const float*)d_in[14]; const float* bg1 = (const float*)d_in[15];
  const float* Wg2 = (const float*)d_in[16]; const float* bg2 = (const float*)d_in[17];
  const float* Wg3 = (const float*)d_in[18]; const float* bg3 = (const float*)d_in[19];
  const float* W_semi  = (const float*)d_in[20];
  const float* bn_g    = (const float*)d_in[21];
  const float* bn_b    = (const float*)d_in[22];
  const float* bn_m    = (const float*)d_in[23];
  const float* bn_v    = (const float*)d_in[24];
  const float* W_final = (const float*)d_in[25];
  const float* b_final = (const float*)d_in[26];

  int N = in_sizes[0];
  int E = in_sizes[1] / 2;
  int G = (out_size - N) / 202;  // out_size = G + G + N + 200*G

  const int* edge_dst = edge_index;      // row 0: scatter targets
  const int* edge_src = edge_index + E;  // row 1: gather sources

  // workspace layout
  char* p = (char*)d_ws;
  _Float16* h0 = (_Float16*)p; p += (size_t)N * 64 * sizeof(_Float16);
  float* h1 = (float*)p;       p += (size_t)N * 64 * sizeof(float);
  float* h2 = (float*)p;       p += (size_t)N * 16 * sizeof(float);
  float* gate = (float*)p;     p += (size_t)N * sizeof(float);
  float* expv = (float*)p;     p += (size_t)N * sizeof(float);
  float* segmax = (float*)p;   p += (size_t)G * sizeof(float);
  float* segsum = (float*)p;   p += (size_t)G * sizeof(float);
  float* emb = (float*)p;      p += (size_t)G * 16 * sizeof(float);
  float* gate_total = (float*)p;

  float* out  = (float*)d_out;
  float* sig  = out + G;
  float* att  = out + 2 * G;
  float* out1 = out + 2 * G + N;

  int ntiles_e = (E + 15) / 16;
  int ntiles_n = (N + 15) / 16;
  int be = (ntiles_e + 7) / 8; if (be > 2048) be = 2048;
  int bn = (ntiles_n + 7) / 8; if (bn > 2048) bn = 2048;

  k_embed<<<(unsigned)(((size_t)N * 64 + 255) / 256), 256, 0, stream>>>(x, embeds, h0, N);
  k_root<64, true><<<(unsigned)bn, 256, 0, stream>>>(h0, W_root1, b_root1, h1, N);
  k_edge<64, true><<<(unsigned)be, 256, 0, stream>>>(h0, edge_dst, edge_src, edge_attr,
                                                     W_neg1, b_neg1, h1, E);
  k_root<16, false><<<(unsigned)bn, 256, 0, stream>>>(h1, W_root2, b_root2, h2, N);
  k_edge<16, false><<<(unsigned)be, 256, 0, stream>>>(h1, edge_dst, edge_src, edge_attr,
                                                      W_neg2, b_neg2, h2, E);
  k_init<<<(unsigned)((G * 16 + 255) / 256), 256, 0, stream>>>(segmax, segsum, emb,
                                                               gate_total, G);
  k_gate<<<(unsigned)((N + 255) / 256), 256, 0, stream>>>(h2, Wg1, bg1, Wg2, bg2,
                                                          Wg3, bg3, gate, gate_total, N);
  k_segmax<<<(unsigned)((N + 255) / 256), 256, 0, stream>>>(gate, batch, segmax, N);
  k_exp<<<(unsigned)((N + 255) / 256), 256, 0, stream>>>(gate, batch, segmax, expv,
                                                         segsum, gate_total, att, N);
  k_scatter<<<(unsigned)(((size_t)N * 16 + 255) / 256), 256, 0, stream>>>(h2, expv, segsum,
                                                                          batch, emb, N);
  k_final<<<(unsigned)((G + 255) / 256), 256, 0, stream>>>(emb, W_semi, bn_g, bn_b, bn_m,
                                                           bn_v, W_final, b_final,
                                                           out, sig, out1, G);
}
